// SelfAttentionPair_44444321579153
// MI455X (gfx1250) — matrix operand order repaired and checked by reading, not compiled, not run on hardware
//
#include <hip/hip_runtime.h>
#include <hip/hip_bf16.h>
#include <cstdint>

typedef __attribute__((ext_vector_type(16))) _Float16 v16h;
typedef __attribute__((ext_vector_type(8)))  float    v8f;
typedef _Float16 half_t;

#define LDS_PITCH 264                 // halfs per row (256 + 8 pad, keeps 16B align)
#define LDS_MAT   (256 * LDS_PITCH)  // halfs per 256x256 buffer
#define LDS_BYTES (2u * LDS_MAT * sizeof(half_t))   // 270336 B < 320 KB

// ---------------------------------------------------------------------------
// B-fragment packing: V_WMMA_F32_16X16X32_F16 B matrix (32x16, KxN), this code's contiguous k order:
//   lane L holds column N = (L&15); half h holds K = h + 16*(L>>4).
// We pre-pack every 32x16 (K-block, N-block) tile so a lane's whole fragment
// is 16 contiguous halfs (32B) -> one vector load per fragment.
__device__ __forceinline__ int pack_index(int k, int n) {
    int kb = k >> 5, kr = k & 31;
    int nb = n >> 4, nr = n & 15;
    int L  = ((kr >> 4) << 4) | nr;
    int h  = kr & 15;
    return ((((kb << 4) | nb) << 5) + L) * 16 + h;   // ((kb*16+nb)*32+L)*16+h
}

// ---------------------------------------------------------------------------
// A-fragment (16x32 f16, this code's contiguous k order): lane L row M = r0 + (L&15);
//   u32 pair i (0..7): K = k0 + 2i + 16*(L>>4)   (matches packed B order)
// -> two contiguous 16B LDS loads per lane.
union AFrag { v16h v; uint4 q[2]; };

__device__ __forceinline__ v16h load_a_frag(const half_t* __restrict__ lds,
                                            int row0, int k0, int lane) {
    AFrag f;
    const half_t* base = lds + (size_t)(row0 + (lane & 15)) * LDS_PITCH
                             + k0 + ((lane >> 4) << 4);
    f.q[0] = *(const uint4*)(base);        // K offsets +0..7
    f.q[1] = *(const uint4*)(base + 8);    // K offsets +8..15
    return f.v;
}

__device__ __forceinline__ v16h load_b_frag(const half_t* __restrict__ pk,
                                            int kb, int nb, int lane) {
    return *(const v16h*)(pk + ((((kb << 4) | nb) << 5) + lane) * 16);
}

// ---------------------------------------------------------------------------
// One 256x256x256 GEMM, C = scale*(A @ B) + bias, A row-major f16 in LDS,
// B packed in global, C written row-major f16 to LDS. Wave w owns rows
// [32w, 32w+32). All 16 A fragments (2 row-tiles x 8 K-steps) are held in
// registers (128 VGPRs) for the whole GEMM; the j loop only streams B.
__device__ __forceinline__ void gemm256(const half_t* __restrict__ A,
                                        const half_t* __restrict__ Bpk,
                                        half_t* __restrict__ C,
                                        int wave, int lane,
                                        float scale, const float* __restrict__ bias) {
    const int r0 = wave << 5;

    v16h a[16];
#pragma unroll
    for (int kk = 0; kk < 8; ++kk) {
        a[2 * kk]     = load_a_frag(A, r0,      kk << 5, lane);
        a[2 * kk + 1] = load_a_frag(A, r0 + 16, kk << 5, lane);
    }

    const int col_in = lane & 15;
    const int rb0    = r0 + ((lane >> 4) << 3);

    for (int j = 0; j < 16; ++j) {
        v8f acc0 = {}; v8f acc1 = {};
#pragma unroll
        for (int kk = 0; kk < 8; ++kk) {
            v16h b = load_b_frag(Bpk, kk, j, lane);
            acc0 = __builtin_amdgcn_wmma_f32_16x16x32_f16(false, a[2 * kk],     false, b,
                                                          (short)0, acc0, false, false);
            acc1 = __builtin_amdgcn_wmma_f32_16x16x32_f16(false, a[2 * kk + 1], false, b,
                                                          (short)0, acc1, false, false);
        }
        // C layout: VGPR r -> row M = r + 8*(lane>>4), col N = lane&15
        const int col = (j << 4) + col_in;
        const float bv = bias ? bias[col] : 0.0f;
#pragma unroll
        for (int r = 0; r < 8; ++r) {
            C[(size_t)(rb0 + r)      * LDS_PITCH + col] = (half_t)(acc0[r] * scale + bv);
            C[(size_t)(rb0 + r + 16) * LDS_PITCH + col] = (half_t)(acc1[r] * scale + bv);
        }
    }
}

// ---------------------------------------------------------------------------
// Prep 1: pack Wk / Wo into B-fragment order (f16).
__global__ __launch_bounds__(256) void pack_weights_kernel(
        const float* __restrict__ Wk, const float* __restrict__ Wo,
        half_t* __restrict__ Wk_pk, half_t* __restrict__ Wo_pk) {
    int tid = blockIdx.x * 256 + threadIdx.x;     // 0..131071
    int which = tid >> 16;
    int p = tid & 65535;
    int h  = p & 15;
    int L  = (p >> 4) & 31;
    int nb = (p >> 9) & 15;
    int kb = p >> 13;
    int k = (kb << 5) + h + ((L >> 4) << 4);
    int n = (nb << 4) + (L & 15);
    const float* W = which ? Wo : Wk;
    half_t* o = which ? Wo_pk : Wk_pk;
    o[p] = (half_t)W[k * 256 + n];
}

// Prep 2: q = hidden@Wq+bq, v = hidden@Wv+bv; store qT (B[k=d][n]=q[n][d])
// and v (B[k=n][d]) in packed fragment order per batch. Tiny: 0.5 GFLOP.
__global__ __launch_bounds__(256) void proj_pack_kernel(
        const float* __restrict__ hidden,
        const float* __restrict__ Wq, const float* __restrict__ bq,
        const float* __restrict__ Wv, const float* __restrict__ bv,
        half_t* __restrict__ qT_pk, half_t* __restrict__ v_pk) {
    int tid = blockIdx.x * 256 + threadIdx.x;     // b*65536 + n*256 + d
    int d = tid & 255, n = (tid >> 8) & 255, b = tid >> 16;
    const float* h = hidden + (size_t)((b << 8) | n) * 256;
    float q = bq[d], v = bv[d];
    for (int k = 0; k < 256; ++k) {
        float hv = h[k];
        q = fmaf(hv, Wq[k * 256 + d], q);
        v = fmaf(hv, Wv[k * 256 + d], v);
    }
    qT_pk[(size_t)b * 65536 + pack_index(d, n)] = (half_t)q;   // scores B = q^T
    v_pk [(size_t)b * 65536 + pack_index(n, d)] = (half_t)v;   // ctx    B = v
}

// ---------------------------------------------------------------------------
// Fused main kernel: one workgroup (256 thr = 8 wave32) per (b,l) tile.
//   LDS A: S_f16 -> scores/probs -> out      LDS C: K_f16 -> ctx_f16
// 270 KB LDS => exactly 1 workgroup/WGP (2 waves/SIMD); launch_bounds(256,1)
// frees the full register file so the 16 A fragments never spill.
__global__ __launch_bounds__(256, 1) void fused_pair_attn_kernel(
        const float* __restrict__ structure,
        const half_t* __restrict__ Wk_pk,
        const float* __restrict__ bk,
        const half_t* __restrict__ qT_pk,
        const half_t* __restrict__ v_pk,
        const half_t* __restrict__ Wo_pk,
        const float* __restrict__ bo,
        const float* __restrict__ gamma,
        const float* __restrict__ beta,
        float* __restrict__ out) {
    extern __shared__ half_t smem[];
    half_t* A = smem;
    half_t* C = smem + LDS_MAT;

    const int tile = blockIdx.x;                 // b*256 + l
    const int b    = tile >> 8;
    const float* Sblk = structure + (size_t)tile * 65536;
    const int lane = threadIdx.x & 31;
    const int wave = threadIdx.x >> 5;

    // ---- stage structure block f32 -> f16 (wave per row, coalesced 1KB rows)
    for (int i = 0; i < 32; ++i) {
        int r = wave + (i << 3);
        const float4* src = (const float4*)(Sblk + (size_t)r * 256) + (lane << 1);
        float4 x0 = src[0], x1 = src[1];
        union { half_t h[8]; uint4 u; } t;
        t.h[0] = (half_t)x0.x; t.h[1] = (half_t)x0.y; t.h[2] = (half_t)x0.z; t.h[3] = (half_t)x0.w;
        t.h[4] = (half_t)x1.x; t.h[5] = (half_t)x1.y; t.h[6] = (half_t)x1.z; t.h[7] = (half_t)x1.w;
        *(uint4*)(A + (size_t)r * LDS_PITCH + (lane << 3)) = t.u;
        if (tile + 1 < (int)gridDim.x)          // warm L2 for the next tile
            __builtin_prefetch(Sblk + 65536 + (size_t)r * 256 + (lane << 3), 0, 1);
    }
    __syncthreads();

    // ---- GEMM1: K = S @ Wk + bk        (C <- A)
    gemm256(A, Wk_pk, C, wave, lane, 1.0f, bk);
    __syncthreads();

    // ---- GEMM2: scores = (K @ q^T)/16  (A <- C)
    gemm256(C, qT_pk + (size_t)b * 65536, A, wave, lane, 0.0625f, nullptr);
    __syncthreads();

    // ---- softmax over n, one row per thread, in place in A
    {
        half_t* row = A + (size_t)threadIdx.x * LDS_PITCH;
        float m = -1e30f;
        for (int i = 0; i < 256; ++i) m = fmaxf(m, (float)row[i]);
        float s = 0.0f;
        for (int i = 0; i < 256; ++i) {
            float e = __expf((float)row[i] - m);
            s += e;
            row[i] = (half_t)e;
        }
        float inv = 1.0f / s;
        for (int i = 0; i < 256; ++i) row[i] = (half_t)((float)row[i] * inv);
    }
    __syncthreads();

    // ---- GEMM3: ctx = probs @ v        (C <- A)
    gemm256(A, v_pk + (size_t)b * 65536, C, wave, lane, 1.0f, nullptr);
    __syncthreads();

    // ---- GEMM4: out = ctx @ Wo + bo    (A <- C)
    gemm256(C, Wo_pk, A, wave, lane, 1.0f, bo);
    __syncthreads();

    // ---- epilogue: layernorm(out + structure) ; wave per row, shuffle reduce
    for (int i = 0; i < 32; ++i) {
        int r = wave + (i << 3);
        const half_t* orow = A + (size_t)r * LDS_PITCH + (lane << 3);
        const float4* srow = (const float4*)(Sblk + (size_t)r * 256) + (lane << 1);
        float4 s0 = srow[0], s1 = srow[1];
        float x[8];
        x[0] = (float)orow[0] + s0.x; x[1] = (float)orow[1] + s0.y;
        x[2] = (float)orow[2] + s0.z; x[3] = (float)orow[3] + s0.w;
        x[4] = (float)orow[4] + s1.x; x[5] = (float)orow[5] + s1.y;
        x[6] = (float)orow[6] + s1.z; x[7] = (float)orow[7] + s1.w;
        float sum = 0.0f, sq = 0.0f;
#pragma unroll
        for (int jj = 0; jj < 8; ++jj) { sum += x[jj]; sq += x[jj] * x[jj]; }
#pragma unroll
        for (int off = 16; off > 0; off >>= 1) {   // wave32 reduction
            sum += __shfl_xor(sum, off, 32);
            sq  += __shfl_xor(sq,  off, 32);
        }
        float mu   = sum * (1.0f / 256.0f);
        float var  = sq * (1.0f / 256.0f) - mu * mu;
        float rstd = rsqrtf(var + 1e-5f);
        const float4* g4 = (const float4*)gamma + (lane << 1);
        const float4* b4 = (const float4*)beta  + (lane << 1);
        float4 g0 = g4[0], g1 = g4[1], e0 = b4[0], e1 = b4[1];
        float4 o0, o1;
        o0.x = (x[0] - mu) * rstd * g0.x + e0.x;
        o0.y = (x[1] - mu) * rstd * g0.y + e0.y;
        o0.z = (x[2] - mu) * rstd * g0.z + e0.z;
        o0.w = (x[3] - mu) * rstd * g0.w + e0.w;
        o1.x = (x[4] - mu) * rstd * g1.x + e1.x;
        o1.y = (x[5] - mu) * rstd * g1.y + e1.y;
        o1.z = (x[6] - mu) * rstd * g1.z + e1.z;
        o1.w = (x[7] - mu) * rstd * g1.w + e1.w;
        float* op = out + (size_t)tile * 65536 + (size_t)r * 256 + (lane << 3);
        *(float4*)op       = o0;
        *((float4*)op + 1) = o1;
    }
}

// ---------------------------------------------------------------------------
extern "C" void kernel_launch(void* const* d_in, const int* in_sizes, int n_in,
                              void* d_out, int out_size, void* d_ws, size_t ws_size,
                              hipStream_t stream) {
    const float* hidden    = (const float*)d_in[0];
    const float* structure = (const float*)d_in[1];
    const float* Wq = (const float*)d_in[2];
    const float* bq = (const float*)d_in[3];
    const float* Wk = (const float*)d_in[4];
    const float* bk = (const float*)d_in[5];
    const float* Wv = (const float*)d_in[6];
    const float* bv = (const float*)d_in[7];
    const float* Wo = (const float*)d_in[8];
    const float* bo = (const float*)d_in[9];
    const float* gamma = (const float*)d_in[10];
    const float* beta  = (const float*)d_in[11];
    float* out = (float*)d_out;

    // workspace layout (halfs): Wk_pk | Wo_pk | qT_pk[8] | v_pk[8]  (~2.36 MB)
    half_t* ws    = (half_t*)d_ws;
    half_t* Wk_pk = ws;
    half_t* Wo_pk = ws + 65536;
    half_t* qT_pk = ws + 131072;
    half_t* v_pk  = ws + 131072 + 8 * 65536;

    (void)hipFuncSetAttribute((const void*)fused_pair_attn_kernel,
                              hipFuncAttributeMaxDynamicSharedMemorySize,
                              (int)LDS_BYTES);

    pack_weights_kernel<<<512, 256, 0, stream>>>(Wk, Wo, Wk_pk, Wo_pk);
    proj_pack_kernel<<<2048, 256, 0, stream>>>(hidden, Wq, bq, Wv, bv, qT_pk, v_pk);
    fused_pair_attn_kernel<<<2048, 256, LDS_BYTES, stream>>>(
        structure, Wk_pk, bk, qT_pk, v_pk, Wo_pk, bo, gamma, beta, out);
}
